// Qwen3VLVisionModel_40827959116130
// MI455X (gfx1250) — matrix operand order repaired and checked by reading, not compiled, not run on hardware
//
#include <hip/hip_runtime.h>
#include <hip/hip_bf16.h>

// ---------------- types ----------------
typedef __attribute__((ext_vector_type(16))) __bf16 v16bf;
typedef __attribute__((ext_vector_type(16))) unsigned short v16u16;
typedef __attribute__((ext_vector_type(8))) float v8f;

#define SEQ 1024
#define DMODEL 1024
#define HEADS 16
#define HD 64
#define FFDIM 4096
#define DEPTH 24
#define OUTD 2560

__device__ __forceinline__ unsigned short f2bf(float f) {
  unsigned int u = __float_as_uint(f);
  unsigned int r = 0x7fffu + ((u >> 16) & 1u);   // round-to-nearest-even
  return (unsigned short)((u + r) >> 16);
}

__device__ __forceinline__ unsigned pack2bf(float lo, float hi) {
  return (unsigned)f2bf(lo) | ((unsigned)f2bf(hi) << 16);
}

__device__ __forceinline__ float gelu_exact(float x) {
  return 0.5f * x * (1.0f + erff(x * 0.7071067811865475f));
}

// low 32 bits of a flat pointer to LDS == LDS byte offset
__device__ __forceinline__ unsigned lds_u32(const void* p) {
  return (unsigned)(reinterpret_cast<unsigned long long>(p));
}

#define BM 128
#define BN 128
#define BK 32
#define LSTR 40   // ushorts per LDS row = 80 B (16B aligned)

// =================================================================
// Weight-path GEMM: C = A(bf16) @ W(bf16,[N,K])^T + bias (+gelu) (+add)
// M,N multiples of 128, K multiple of 32 -> no bounds checks.
// BOTH tiles: double-buffered global_load_async_to_lds_b128 (ASYNCcnt DMA),
// no conversion VALU and no VGPR staging in the inner loop.
// =================================================================
__global__ void __launch_bounds__(256)
gemm_wt_bf16(const unsigned short* __restrict__ Abf, long long lda,
             const unsigned short* __restrict__ Bw, long long ldb,
             void* __restrict__ Cv, long long ldc,
             const float* __restrict__ bias,
             const float* __restrict__ add, long long ldadd,
             int M, int N, int K, int doGelu, int outBf16)
{
  __shared__ unsigned short As[2][BM * LSTR];
  __shared__ unsigned short Bs[2][BN * LSTR];
  const int tid = threadIdx.x;
  const int tileM = blockIdx.y * BM;
  const int tileN = blockIdx.x * BN;

  const int wave = tid >> 5;
  const int lane = tid & 31;
  const int wm = wave & 3;        // 4 waves along M
  const int wn = wave >> 2;       // 2 waves along N
  const int lrow = lane & 15;
  const int lsel = lane >> 4;
  const int ka = lsel * 8;
  const int kb = lsel * 16;

  auto loadA = [&](int k0, int buf) {
#pragma unroll
    for (int it = 0; it < 2; ++it) {
      int id = it * 256 + tid;        // 512 x 16B chunks = 128x32 bf16 tile
      int m = id >> 2;
      int kk = (id & 3) << 3;
      const unsigned short* src = Abf + (long long)(tileM + m) * lda + (k0 + kk);
      unsigned dst = lds_u32(&As[buf][m * LSTR + kk]);
      asm volatile("global_load_async_to_lds_b128 %0, %1, off"
                   :: "v"(dst), "v"((unsigned long long)src) : "memory");
    }
  };
  auto loadB = [&](int k0, int buf) {
#pragma unroll
    for (int it = 0; it < 2; ++it) {
      int id = it * 256 + tid;
      int n = id >> 2;
      int kk = (id & 3) << 3;
      const unsigned short* src = Bw + (long long)(tileN + n) * ldb + (k0 + kk);
      unsigned dst = lds_u32(&Bs[buf][n * LSTR + kk]);
      asm volatile("global_load_async_to_lds_b128 %0, %1, off"
                   :: "v"(dst), "v"((unsigned long long)src) : "memory");
    }
  };

  v8f acc[2][4];
#pragma unroll
  for (int i = 0; i < 2; ++i)
#pragma unroll
    for (int j = 0; j < 4; ++j)
#pragma unroll
      for (int e = 0; e < 8; ++e) acc[i][j][e] = 0.0f;

  loadB(0, 0);
  loadA(0, 0);
  asm volatile("s_wait_asynccnt 0x0" ::: "memory");
  __syncthreads();

  int cur = 0;
  for (int k0 = 0; k0 < K; k0 += BK) {
    int nxt = cur ^ 1;
    if (k0 + BK < K) {                 // prefetch next tiles into other buffer
      loadB(k0 + BK, nxt);
      loadA(k0 + BK, nxt);
    }
    const unsigned short* as = As[cur];
    const unsigned short* bs = Bs[cur];

    v16bf af[2], bfm[4];
#pragma unroll
    for (int tm = 0; tm < 2; ++tm) {
      int row = wm * 32 + tm * 16 + lrow;
      v16u16 u;
#pragma unroll
      for (int j = 0; j < 8; ++j) {
        u[j]     = as[row * LSTR + ka + j];
        u[8 + j] = as[row * LSTR + 16 + ka + j];
      }
      af[tm] = __builtin_bit_cast(v16bf, u);
    }
#pragma unroll
    for (int tn = 0; tn < 4; ++tn) {
      int col = wn * 64 + tn * 16 + lrow;
      v16u16 u;
#pragma unroll
      for (int j = 0; j < 8; ++j) {
        u[j]     = bs[col * LSTR + ka + j];
        u[8 + j] = bs[col * LSTR + 16 + ka + j];
      }
      bfm[tn] = __builtin_bit_cast(v16bf, u);
    }
#pragma unroll
    for (int tm = 0; tm < 2; ++tm)
#pragma unroll
      for (int tn = 0; tn < 4; ++tn)
        acc[tm][tn] = __builtin_amdgcn_wmma_f32_16x16x32_bf16(
            false, af[tm], false, bfm[tn], (short)0, acc[tm][tn], false, false);

    asm volatile("s_wait_asynccnt 0x0" ::: "memory");
    __syncthreads();
    cur = nxt;
  }

#pragma unroll
  for (int tm = 0; tm < 2; ++tm) {
#pragma unroll
    for (int tn = 0; tn < 4; ++tn) {
      int col = tileN + wn * 64 + tn * 16 + lrow;
      float bv = bias[col];
#pragma unroll
      for (int r = 0; r < 8; ++r) {
        int row = tileM + wm * 32 + tm * 16 + lsel * 8 + r;
        float v = acc[tm][tn][r] + bv;
        if (doGelu) v = gelu_exact(v);
        if (add) v += add[(long long)row * ldadd + col];
        if (outBf16) ((unsigned short*)Cv)[(long long)row * ldc + col] = f2bf(v);
        else         ((float*)Cv)[(long long)row * ldc + col] = v;
      }
    }
  }
}

// =================================================================
// Generic fp32-operand GEMM (attention): C = A @ op(B), batched
// bTrans=1 -> B [N,K]; bTrans=0 -> B [K,N]. Bounds-checked.
// =================================================================
__global__ void __launch_bounds__(256)
gemm_wmma_f32(const float* __restrict__ A, long long lda, long long sA,
              const float* __restrict__ B, long long ldb, long long sB,
              void* __restrict__ Cv, long long ldc, long long sC,
              int M, int N, int K, int bTrans, int outBf16)
{
  __shared__ unsigned short As[BM * LSTR];
  __shared__ unsigned short Bs[BN * LSTR];
  const int tid = threadIdx.x;
  const int bz = blockIdx.z;
  const int tileM = blockIdx.y * BM;
  const int tileN = blockIdx.x * BN;
  A += (long long)bz * sA;
  B += (long long)bz * sB;

  const int wave = tid >> 5;
  const int lane = tid & 31;
  const int wm = wave & 3;
  const int wn = wave >> 2;
  const int lrow = lane & 15;
  const int lsel = lane >> 4;
  const int ka = lsel * 8;
  const int kb = lsel * 16;

  v8f acc[2][4];
#pragma unroll
  for (int i = 0; i < 2; ++i)
#pragma unroll
    for (int j = 0; j < 4; ++j)
#pragma unroll
      for (int e = 0; e < 8; ++e) acc[i][j][e] = 0.0f;

  for (int k0 = 0; k0 < K; k0 += BK) {
    {
      float4 v[4];
#pragma unroll
      for (int it = 0; it < 4; ++it) {
        int id = it * 256 + tid;
        int m = id >> 3;
        int kk = (id & 7) << 2;
        int gm = tileM + m;
        v[it] = make_float4(0.f, 0.f, 0.f, 0.f);
        if (gm < M) v[it] = *(const float4*)(A + (long long)gm * lda + (k0 + kk));
      }
#pragma unroll
      for (int it = 0; it < 4; ++it) {
        int id = it * 256 + tid;
        int m = id >> 3;
        int kk = (id & 7) << 2;
        unsigned short* d = &As[m * LSTR + kk];
        d[0] = f2bf(v[it].x); d[1] = f2bf(v[it].y);
        d[2] = f2bf(v[it].z); d[3] = f2bf(v[it].w);
      }
    }
    if (bTrans) {
      float4 v[4];
#pragma unroll
      for (int it = 0; it < 4; ++it) {
        int id = it * 256 + tid;
        int n = id >> 3;
        int kk = (id & 7) << 2;
        int gn = tileN + n;
        v[it] = make_float4(0.f, 0.f, 0.f, 0.f);
        if (gn < N) v[it] = *(const float4*)(B + (long long)gn * ldb + (k0 + kk));
      }
#pragma unroll
      for (int it = 0; it < 4; ++it) {
        int id = it * 256 + tid;
        int n = id >> 3;
        int kk = (id & 7) << 2;
        unsigned short* d = &Bs[n * LSTR + kk];
        d[0] = f2bf(v[it].x); d[1] = f2bf(v[it].y);
        d[2] = f2bf(v[it].z); d[3] = f2bf(v[it].w);
      }
    } else {
      float v[16];
#pragma unroll
      for (int it = 0; it < 16; ++it) {
        int id = it * 256 + tid;
        int k = id >> 7;
        int n = id & 127;
        int gn = tileN + n;
        v[it] = 0.f;
        if (gn < N) v[it] = B[(long long)(k0 + k) * ldb + gn];
      }
#pragma unroll
      for (int it = 0; it < 16; ++it) {
        int id = it * 256 + tid;
        int k = id >> 7;
        int n = id & 127;
        Bs[n * LSTR + k] = f2bf(v[it]);
      }
    }
    __syncthreads();

    v16bf af[2], bfm[4];
#pragma unroll
    for (int tm = 0; tm < 2; ++tm) {
      int row = wm * 32 + tm * 16 + lrow;
      v16u16 u;
#pragma unroll
      for (int j = 0; j < 8; ++j) {
        u[j]     = As[row * LSTR + ka + j];
        u[8 + j] = As[row * LSTR + 16 + ka + j];
      }
      af[tm] = __builtin_bit_cast(v16bf, u);
    }
#pragma unroll
    for (int tn = 0; tn < 4; ++tn) {
      int col = wn * 64 + tn * 16 + lrow;
      v16u16 u;
#pragma unroll
      for (int j = 0; j < 8; ++j) {
        u[j]     = Bs[col * LSTR + ka + j];
        u[8 + j] = Bs[col * LSTR + 16 + ka + j];
      }
      bfm[tn] = __builtin_bit_cast(v16bf, u);
    }
#pragma unroll
    for (int tm = 0; tm < 2; ++tm)
#pragma unroll
      for (int tn = 0; tn < 4; ++tn)
        acc[tm][tn] = __builtin_amdgcn_wmma_f32_16x16x32_bf16(
            false, af[tm], false, bfm[tn], (short)0, acc[tm][tn], false, false);
    __syncthreads();
  }

#pragma unroll
  for (int tm = 0; tm < 2; ++tm) {
#pragma unroll
    for (int tn = 0; tn < 4; ++tn) {
      int col = tileN + wn * 64 + tn * 16 + lrow;
      if (col >= N) continue;
#pragma unroll
      for (int r = 0; r < 8; ++r) {
        int row = tileM + wm * 32 + tm * 16 + lsel * 8 + r;
        if (row >= M) continue;
        float v = acc[tm][tn][r];
        if (outBf16)
          ((unsigned short*)Cv)[(long long)bz * sC + (long long)row * ldc + col] = f2bf(v);
        else
          ((float*)Cv)[(long long)bz * sC + (long long)row * ldc + col] = v;
      }
    }
  }
}

// ---------------- fp32 -> bf16 conversion (n multiple of 8) ----------------
__global__ void __launch_bounds__(256)
cvt_bf16_kernel(const float* __restrict__ src, unsigned short* __restrict__ dst,
                long long n)
{
  long long i = ((long long)blockIdx.x * 256 + threadIdx.x) * 8;
  if (i + 8 > n) return;
  float4 a = *(const float4*)(src + i);
  float4 b = *(const float4*)(src + i + 4);
  uint4 o;
  o.x = pack2bf(a.x, a.y);
  o.y = pack2bf(a.z, a.w);
  o.z = pack2bf(b.x, b.y);
  o.w = pack2bf(b.z, b.w);
  *(uint4*)(dst + i) = o;
}

// ---------------- LayerNorm (row-wise, biased var), bf16 output ----------------
__global__ void __launch_bounds__(256)
ln_kernel(const float* __restrict__ x, unsigned short* __restrict__ y,
          const float* __restrict__ sc, const float* __restrict__ bi, int cols)
{
  const long long row = blockIdx.x;
  const float* xr = x + row * cols;
  unsigned short* yr = y + row * cols;
  __shared__ float rs[8], rq[8], mv[2];
  float s = 0.f, q = 0.f;
  for (int i = threadIdx.x; i < cols; i += 256) { float v = xr[i]; s += v; q += v * v; }
#pragma unroll
  for (int o = 16; o > 0; o >>= 1) { s += __shfl_xor(s, o, 32); q += __shfl_xor(q, o, 32); }
  int w = threadIdx.x >> 5, l = threadIdx.x & 31;
  if (l == 0) { rs[w] = s; rq[w] = q; }
  __syncthreads();
  if (threadIdx.x == 0) {
    float ts = 0.f, tq = 0.f;
    for (int i = 0; i < 8; ++i) { ts += rs[i]; tq += rq[i]; }
    float mean = ts / cols;
    float var = tq / cols - mean * mean;
    mv[0] = mean; mv[1] = rsqrtf(var + 1e-6f);
  }
  __syncthreads();
  float mean = mv[0], inv = mv[1];
  for (int i = threadIdx.x; i < cols; i += 256)
    yr[i] = f2bf((xr[i] - mean) * inv * sc[i] + bi[i]);
}

// ---------------- scaled softmax over rows ----------------
__global__ void __launch_bounds__(256)
softmax_kernel(float* __restrict__ p, int cols, float scale)
{
  const long long row = blockIdx.x;
  float* r = p + row * cols;
  __shared__ float red[8], bval;
  float m = -3.4e38f;
  for (int i = threadIdx.x; i < cols; i += 256) m = fmaxf(m, r[i] * scale);
#pragma unroll
  for (int o = 16; o; o >>= 1) m = fmaxf(m, __shfl_xor(m, o, 32));
  int w = threadIdx.x >> 5, l = threadIdx.x & 31;
  if (!l) red[w] = m;
  __syncthreads();
  if (threadIdx.x == 0) { float t = red[0]; for (int i = 1; i < 8; ++i) t = fmaxf(t, red[i]); bval = t; }
  __syncthreads();
  m = bval;
  float s = 0.f;
  for (int i = threadIdx.x; i < cols; i += 256) { float e = __expf(r[i] * scale - m); r[i] = e; s += e; }
#pragma unroll
  for (int o = 16; o; o >>= 1) s += __shfl_xor(s, o, 32);
  if (!l) red[w] = s;
  __syncthreads();
  if (threadIdx.x == 0) { float t = 0.f; for (int i = 0; i < 8; ++i) t += red[i]; bval = 1.f / t; }
  __syncthreads();
  float invs = bval;
  for (int i = threadIdx.x; i < cols; i += 256) r[i] *= invs;
}

// ---------------- 2-axis rope applied in-place to packed q,k of qkv ----------------
__global__ void __launch_bounds__(256)
rope_kernel(float* __restrict__ qkv)
{
  const int s = blockIdx.x;
  const int wp = s & 1, hp = (s >> 1) & 1, wb = (s >> 2) & 15, hb = (s >> 6) & 15;
  const float hpos = (float)(hb * 2 + hp);
  const float wpos = (float)(wb * 2 + wp);
#pragma unroll
  for (int it = 0; it < 4; ++it) {
    int id = it * 256 + threadIdx.x;
    int which = id >> 9;
    int pp = id & 511;
    int head = pp >> 5;
    int dp = pp & 31;
    float pos = (dp < 16) ? hpos : wpos;
    int j = dp & 15;
    float inv = __powf(10000.f, -(float)j * (1.f / 16.f));
    float sn, cs;
    __sincosf(pos * inv, &sn, &cs);
    float* base = qkv + (long long)s * 3072 + which * 1024 + head * 64 + dp;
    float x1 = base[0], x2 = base[32];
    base[0]  = x1 * cs - x2 * sn;
    base[32] = x2 * cs + x1 * sn;
  }
}

// ---------------- learned position embedding gather-add ----------------
__global__ void __launch_bounds__(256)
posadd_kernel(float* __restrict__ x, const float* __restrict__ pt)
{
  const int s = blockIdx.x;
  const int wp = s & 1, hp = (s >> 1) & 1, wb = (s >> 2) & 15, hb = (s >> 6) & 15;
  const int h = hb * 2 + hp, w = wb * 2 + wp;
  const float* p = pt + (long long)(h * 48 + w) * 1024;
  float* xr = x + (long long)s * 1024;
  for (int i = threadIdx.x; i < 1024; i += 256) xr[i] += p[i];
}

// ---------------- host side ----------------
static void wt_gemm(hipStream_t st, const unsigned short* A, long long lda,
                    const float* W, int N, int K,
                    void* C, long long ldc, const float* bias,
                    const float* add, long long ldadd, int M,
                    int doGelu, int outBf16, unsigned short* stage)
{
  long long nelem = (long long)N * K;
  cvt_bf16_kernel<<<(unsigned)(nelem / 2048), 256, 0, st>>>(W, stage, nelem);
  dim3 grid(N / BN, M / BM, 1);
  gemm_wt_bf16<<<grid, 256, 0, st>>>(A, lda, stage, K, C, ldc, bias,
                                     add, ldadd, M, N, K, doGelu, outBf16);
}

extern "C" void kernel_launch(void* const* d_in, const int* in_sizes, int n_in,
                              void* d_out, int out_size, void* d_ws, size_t ws_size,
                              hipStream_t stream)
{
  (void)in_sizes; (void)n_in; (void)out_size; (void)ws_size;
  const float* pixel_values = (const float*)d_in[0];
  const float* conv_w   = (const float*)d_in[1];
  const float* conv_b   = (const float*)d_in[2];
  const float* pos_table = (const float*)d_in[3];
  const float* ln1_s = (const float*)d_in[4];
  const float* ln1_b = (const float*)d_in[5];
  const float* ln2_s = (const float*)d_in[6];
  const float* ln2_b = (const float*)d_in[7];
  const float* qkv_w = (const float*)d_in[8];
  const float* qkv_b = (const float*)d_in[9];
  const float* proj_w = (const float*)d_in[10];
  const float* proj_b = (const float*)d_in[11];
  const float* fc1_w = (const float*)d_in[12];
  const float* fc1_b = (const float*)d_in[13];
  const float* fc2_w = (const float*)d_in[14];
  const float* fc2_b = (const float*)d_in[15];
  const float* mrg_ln_s = (const float*)d_in[16];
  const float* mrg_ln_b = (const float*)d_in[17];
  const float* mrg_fc1_w = (const float*)d_in[18];
  const float* mrg_fc1_b = (const float*)d_in[19];
  const float* mrg_fc2_w = (const float*)d_in[20];
  const float* mrg_fc2_b = (const float*)d_in[21];
  const float* ds_ln_s = (const float*)d_in[22];
  const float* ds_ln_b = (const float*)d_in[23];
  const float* ds_fc1_w = (const float*)d_in[24];
  const float* ds_fc1_b = (const float*)d_in[25];
  const float* ds_fc2_w = (const float*)d_in[26];
  const float* ds_fc2_b = (const float*)d_in[27];

  float* ws = (float*)d_ws;
  float* x      = ws; ws += 1048576;                       // [1024,1024] fp32 residual
  unsigned short* h = (unsigned short*)ws; ws += 1048576;  // LN out, bf16 [1024,1024]/[256,4096]
  float* qkv    = ws; ws += 3145728;                       // [1024,3072] fp32 (rope)
  float* scores = ws; ws += 16777216;                      // [16,1024,1024] fp32
  unsigned short* o  = (unsigned short*)ws; ws += 1048576; // attn out, bf16 [1024,1024]
  unsigned short* h1 = (unsigned short*)ws; ws += 4194304; // fc1 out bf16 / px staging
  float* dsbuf[3] = { ws, ws + 1048576, ws + 2097152 }; ws += 3145728;
  unsigned short* stage = (unsigned short*)ws;             // bf16 weights (<=32MB)

  // patch embed: convert pixels to bf16 (into h1 region), per-token linear, pos embed
  cvt_bf16_kernel<<<768, 256, 0, stream>>>(pixel_values, h1, (long long)SEQ * 1536);
  wt_gemm(stream, h1, 1536, conv_w, DMODEL, 1536,
          x, 1024, conv_b, nullptr, 0, SEQ, 0, 0, stage);
  posadd_kernel<<<SEQ, 256, 0, stream>>>(x, pos_table);

  int dsj = 0;
  for (int i = 0; i < DEPTH; ++i) {
    ln_kernel<<<SEQ, 256, 0, stream>>>(x, h, ln1_s + i * 1024, ln1_b + i * 1024, 1024);
    wt_gemm(stream, h, 1024, qkv_w + (long long)i * 3072 * 1024, 3072, 1024,
            qkv, 3072, qkv_b + (long long)i * 3072, nullptr, 0, SEQ, 0, 0, stage);
    rope_kernel<<<SEQ, 256, 0, stream>>>(qkv);
    // S = Q @ K^T per head (batched via z), fp32 out
    {
      dim3 g(SEQ / BN, SEQ / BM, HEADS);
      gemm_wmma_f32<<<g, 256, 0, stream>>>(qkv, 3072, 64, qkv + 1024, 3072, 64,
                                           scores, 1024, 1048576, SEQ, SEQ, 64, 1, 0);
    }
    softmax_kernel<<<HEADS * SEQ, 256, 0, stream>>>(scores, SEQ, 0.125f);
    // O = P @ V per head (B stored [K,N] row stride 3072), bf16 out
    {
      dim3 g(1, SEQ / BM, HEADS);
      gemm_wmma_f32<<<g, 256, 0, stream>>>(scores, 1024, 1048576, qkv + 2048, 3072, 64,
                                           o, 1024, 64, SEQ, HD, SEQ, 0, 1);
    }
    // x = x + O @ projW^T + b
    wt_gemm(stream, o, 1024, proj_w + (long long)i * 1024 * 1024, 1024, 1024,
            x, 1024, proj_b + i * 1024, x, 1024, SEQ, 0, 0, stage);
    ln_kernel<<<SEQ, 256, 0, stream>>>(x, h, ln2_s + i * 1024, ln2_b + i * 1024, 1024);
    wt_gemm(stream, h, 1024, fc1_w + (long long)i * 4096 * 1024, 4096, 1024,
            h1, 4096, fc1_b + (long long)i * 4096, nullptr, 0, SEQ, 1, 1, stage);
    wt_gemm(stream, h1, 4096, fc2_w + (long long)i * 1024 * 4096, 1024, 4096,
            x, 1024, fc2_b + i * 1024, x, 1024, SEQ, 0, 0, stage);
    if (i == 5 || i == 11 || i == 17) {
      hipMemcpyAsync(dsbuf[dsj], x, 1048576 * sizeof(float),
                     hipMemcpyDeviceToDevice, stream);
      ++dsj;
    }
  }

  float* out = (float*)d_out;
  hipMemcpyAsync(out, x, 1048576 * sizeof(float), hipMemcpyDeviceToDevice, stream);

  // pooled merger: LN over D, then view [256,4096]
  ln_kernel<<<SEQ, 256, 0, stream>>>(x, h, mrg_ln_s, mrg_ln_b, 1024);
  wt_gemm(stream, h, 4096, mrg_fc1_w, 4096, 4096, h1, 4096,
          mrg_fc1_b, nullptr, 0, 256, 1, 1, stage);
  wt_gemm(stream, h1, 4096, mrg_fc2_w, 2560, 4096, out + 1048576, 2560,
          mrg_fc2_b, nullptr, 0, 256, 0, 0, stage);

  // deepstack mergers: postshuffle LN over 4D rows
  for (int j = 0; j < 3; ++j) {
    ln_kernel<<<256, 256, 0, stream>>>(dsbuf[j], h, ds_ln_s + j * 4096, ds_ln_b + j * 4096, 4096);
    wt_gemm(stream, h, 4096, ds_fc1_w + (long long)j * 4096 * 4096, 4096, 4096,
            h1, 4096, ds_fc1_b + j * 4096, nullptr, 0, 256, 1, 1, stage);
    wt_gemm(stream, h1, 4096, ds_fc2_w + (long long)j * 2560 * 4096, 2560, 4096,
            out + 1048576 + (long long)(j + 1) * 655360, 2560,
            ds_fc2_b + j * 2560, nullptr, 0, 256, 0, 0, stage);
  }
}
